// ADDLossSoftEncode_52312701665736
// MI455X (gfx1250) — matrix operand order repaired and checked by reading, not compiled, not run on hardware
//
#include <hip/hip_runtime.h>
#include <math.h>
#include <stdint.h>

typedef __attribute__((ext_vector_type(16))) _Float16 v16h;
typedef __attribute__((ext_vector_type(8)))  _Float16 v8h;
typedef __attribute__((ext_vector_type(8)))  float    v8f;
typedef __attribute__((ext_vector_type(4)))  unsigned int v4u;
typedef __attribute__((ext_vector_type(8)))  int      v8i;
typedef __attribute__((ext_vector_type(4)))  int      v4i;

#define BATCH 512
#define K_TOT 8192
#define NPTS  4096
#define KQ    1024              // K-slice per TDM tile
#define NQ    (K_TOT / KQ)      // 8 slices
#define ROWP  (KQ + 8)          // padded row (halfs) to de-alias LDS banks

__device__ __forceinline__ void quat_to_mat(const float q[4], float R[9]) {
    // matches reference quaternion_to_matrix (two_s = 2/|q|^2)
    float r = q[0], i = q[1], j = q[2], k = q[3];
    float s = 2.0f / (r*r + i*i + j*j + k*k);
    R[0] = 1.f - s*(j*j + k*k); R[1] = s*(i*j - k*r);       R[2] = s*(i*k + j*r);
    R[3] = s*(i*j + k*r);       R[4] = 1.f - s*(i*i + k*k); R[5] = s*(j*k - i*r);
    R[6] = s*(i*k - j*r);       R[7] = s*(j*k + i*r);       R[8] = 1.f - s*(i*i + j*j);
}

// Issue a 1-D TDM tile: nelem f32 elements, global -> LDS (async, TENSORcnt).
__device__ __forceinline__ void tdm_load_1d(unsigned lds_off, const void* gptr,
                                            unsigned nelem) {
    unsigned long long ga = (unsigned long long)(uintptr_t)gptr;
    v4u g0;
    g0[0] = 1u;                                              // count=1 (valid user D#)
    g0[1] = lds_off;                                         // lds_addr
    g0[2] = (unsigned)(ga & 0xFFFFFFFFu);                    // global_addr[31:0]
    g0[3] = (unsigned)((ga >> 32) & 0x01FFFFFFu) | 0x80000000u; // ga[56:32] | type=2
    v8i g1;
    g1[0] = (int)(2u << 16);                                 // data_size=2 -> 4 bytes
    g1[1] = (int)((nelem & 0xFFFFu) << 16);                  // tensor_dim0[15:0]
    g1[2] = (int)(((nelem >> 16) & 0xFFFFu) | (1u << 16));   // tensor_dim0[31:16] | tensor_dim1=1
    g1[3] = (int)((nelem & 0xFFFFu) << 16);                  // tile_dim0 = nelem
    g1[4] = 1;                                               // tile_dim1=1, tile_dim2=0
    g1[5] = (int)nelem;                                      // tensor_dim0_stride
    g1[6] = 0;
    g1[7] = 0;
    v4i z = {0, 0, 0, 0};
#if __clang_major__ >= 23
    v8i z8 = {0, 0, 0, 0, 0, 0, 0, 0};
    __builtin_amdgcn_tensor_load_to_lds(g0, g1, z, z, z8, 0);
#else
    __builtin_amdgcn_tensor_load_to_lds(g0, g1, z, z, 0);
#endif
}

// Kernel 1: per batch -- TDM-staged, softmax-weighted Gram via WMMA,
// power iteration, quaternion->rotation, M = Rd*Rd^T -> workspace.
__global__ __launch_bounds__(256)
void gram_eig_kernel(const float* __restrict__ s,
                     const float* __restrict__ h,
                     const float* __restrict__ gt,
                     float* __restrict__ Mout)
{
    __shared__ __align__(16) float    hstage[2][KQ * 4];   // 2 x 16KB TDM double buffer
    __shared__ __align__(16) _Float16 xh[4 * ROWP];        // transposed f16 x-matrix
    __shared__ float red[256];
    __shared__ float gpart[8][16];
    __shared__ float G[16];
    __shared__ float smax_sh;

    const int b    = blockIdx.x;
    const int tid  = threadIdx.x;
    const int lane = tid & 31;
    const int wave = tid >> 5;

    const float* sb = s + (size_t)b * K_TOT;
    const float* hb = h + (size_t)b * K_TOT * 4;

    // ---- pass 1: max over s[b,:]  (2nd read of s later hits the 192MB L2)
    float mx = -3.4e38f;
    for (int k = tid; k < K_TOT; k += 256)
        mx = fmaxf(mx, sb[k]);
    red[tid] = mx;
    __syncthreads();
    for (int off = 128; off > 0; off >>= 1) {
        if (tid < off) red[tid] = fmaxf(red[tid], red[tid + off]);
        __syncthreads();
    }
    if (tid == 0) smax_sh = red[0];
    __syncthreads();
    const float smax = smax_sh;

    const unsigned lds_buf0 = (unsigned)(unsigned long long)(uintptr_t)(void*)&hstage[0][0];
    const unsigned lds_buf1 = (unsigned)(unsigned long long)(uintptr_t)(void*)&hstage[1][0];

    // ---- prime the TDM pipeline: slice 0 -> buffer 0
    if (wave == 0)
        tdm_load_1d(lds_buf0, hb, KQ * 4);

    v8f  acc = {};            // f32 Gram accumulator (top-left 4x4 of C used)
    v16h afr = {}, bfr = {};  // lanes with m>=4 keep zeros forever
    const int m = lane & 15;

    for (int q = 0; q < NQ; ++q) {
        // ---- issue next slice, then wait for current slice's DMA
        if (wave == 0) {
            if (q + 1 < NQ) {
                tdm_load_1d(((q + 1) & 1) ? lds_buf1 : lds_buf0,
                            hb + (size_t)(q + 1) * KQ * 4, KQ * 4);
                __builtin_amdgcn_s_wait_tensorcnt(1);   // current slice done
            } else {
                __builtin_amdgcn_s_wait_tensorcnt(0);
            }
        }
        __syncthreads();   // publish TDM-written LDS to all waves

        // ---- convert: x_k = sqrt(exp(s_k - max)) * h_k  (f32 LDS -> f16 LDS^T)
        {
            const float4* hs4 = (const float4*)&hstage[q & 1][0];
            const float*  sq  = sb + q * KQ;
            for (int k = tid; k < KQ; k += 256) {
                float r = sqrtf(__expf(sq[k] - smax));
                float4 hv = hs4[k];
                xh[0 * ROWP + k] = (_Float16)(r * hv.x);
                xh[1 * ROWP + k] = (_Float16)(r * hv.y);
                xh[2 * ROWP + k] = (_Float16)(r * hv.z);
                xh[3 * ROWP + k] = (_Float16)(r * hv.w);
            }
        }
        __syncthreads();

        // ---- WMMA Gram accumulation over this slice (32 chunks of 32 k)
        for (int c = wave; c < KQ / 32; c += 8) {
            const int k0 = c * 32;
            if (m < 4) {
                const _Float16* row = xh + m * ROWP;
                // A 16x32 f16 layout: lanes 0-15: K0-7 / K16-23; lanes 16-31: K8-15 / K24-31
                const int ka = (lane < 16) ? 0 : 8;
                v8h alo = *(const v8h*)(row + k0 + ka);
                v8h ahi = *(const v8h*)(row + k0 + ka + 16);
                afr = __builtin_shufflevector(alo, ahi, 0,1,2,3,4,5,6,7,
                                                        8,9,10,11,12,13,14,15);
                // B 32x16 f16 layout: lanes 0-15: K0-7 / K16-23; lanes 16-31: K8-15 / K24-31 (col n)
                const int kb = (lane < 16) ? 0 : 8;
                v8h blo = *(const v8h*)(row + k0 + kb);
                v8h bhi = *(const v8h*)(row + k0 + kb + 16);
                bfr = __builtin_shufflevector(blo, bhi, 0,1,2,3,4,5,6,7,
                                                        8,9,10,11,12,13,14,15);
            }
            acc = __builtin_amdgcn_wmma_f32_16x16x32_f16(
                      false, afr, false, bfr, (short)0, acc, false, false);
        }
        __syncthreads();   // protect xh before next slice overwrites it
    }

    // ---- gather top-left 4x4 of each wave's C: lane n(0..3), VGPR r -> C[r][n]
    if (lane < 4) {
        for (int r = 0; r < 4; ++r)
            gpart[wave][r * 4 + lane] = acc[r];
    }
    __syncthreads();
    if (tid < 16) {
        float g = 0.f;
        for (int w = 0; w < 8; ++w) g += gpart[w][tid];
        G[tid] = g;
    }
    __syncthreads();

    if (tid == 0) {
        // power iteration: G is PSD -> dominant eigenvector == eigh(...)[:, -1];
        // sign ambiguity irrelevant (q and -q give the same rotation)
        float q[4] = {0.5f, 0.5f, 0.5f, 0.5f};
        for (int it = 0; it < 64; ++it) {
            float t[4];
            for (int i = 0; i < 4; ++i)
                t[i] = G[i*4+0]*q[0] + G[i*4+1]*q[1] + G[i*4+2]*q[2] + G[i*4+3]*q[3];
            float inv = rsqrtf(t[0]*t[0] + t[1]*t[1] + t[2]*t[2] + t[3]*t[3] + 1e-30f);
            for (int i = 0; i < 4; ++i) q[i] = t[i] * inv;
        }
        float Rp[9], Rg[9], Rd[9];
        quat_to_mat(q, Rp);
        float qg[4] = {gt[b*4+0], gt[b*4+1], gt[b*4+2], gt[b*4+3]};
        quat_to_mat(qg, Rg);
        for (int t = 0; t < 9; ++t) Rd[t] = Rp[t] - Rg[t];
        for (int i = 0; i < 3; ++i)           // M = Rd * Rd^T
            for (int j = 0; j < 3; ++j)
                Mout[b*9 + i*3 + j] =
                    Rd[i*3+0]*Rd[j*3+0] + Rd[i*3+1]*Rd[j*3+1] + Rd[i*3+2]*Rd[j*3+2];
    }
}

// Kernel 2: per batch, sum_p sqrt(p^T M p)  -> block partial
__global__ __launch_bounds__(256)
void point_loss_kernel(const float* __restrict__ point,
                       const float* __restrict__ Mw,
                       float* __restrict__ partial)
{
    __shared__ float red[256];
    __shared__ float Ms[9];
    const int b   = blockIdx.x;
    const int tid = threadIdx.x;
    if (tid < 9) Ms[tid] = Mw[b*9 + tid];
    __syncthreads();
    const float M00 = Ms[0], M01 = Ms[1], M02 = Ms[2];
    const float M11 = Ms[4], M12 = Ms[5], M22 = Ms[8];

    const float* pb = point + (size_t)b * NPTS * 3;
    float sum = 0.f;
    for (int p = tid; p < NPTS; p += 256) {
        if (p + 512 < NPTS)
            __builtin_prefetch(pb + (p + 512) * 3, 0, 1);   // global_prefetch_b8
        float x = pb[p*3 + 0], y = pb[p*3 + 1], z = pb[p*3 + 2];
        float qf = M00*x*x + M11*y*y + M22*z*z
                 + 2.f*(M01*x*y + M02*x*z + M12*y*z);
        sum += sqrtf(fmaxf(qf, 0.f));
    }
    red[tid] = sum;
    __syncthreads();
    for (int off = 128; off > 0; off >>= 1) {
        if (tid < off) red[tid] += red[tid + off];
        __syncthreads();
    }
    if (tid == 0) partial[b] = red[0];
}

// Kernel 3: deterministic final reduce + mean (no atomics, no d_out init needed)
__global__ __launch_bounds__(512)
void finalize_kernel(const float* __restrict__ partial, float* __restrict__ out)
{
    __shared__ float red[512];
    const int tid = threadIdx.x;
    red[tid] = partial[tid];
    __syncthreads();
    for (int off = 256; off > 0; off >>= 1) {
        if (tid < off) red[tid] += red[tid + off];
        __syncthreads();
    }
    if (tid == 0) out[0] = red[0] * (1.0f / ((float)BATCH * (float)NPTS));
}

extern "C" void kernel_launch(void* const* d_in, const int* in_sizes, int n_in,
                              void* d_out, int out_size, void* d_ws, size_t ws_size,
                              hipStream_t stream) {
    const float* s     = (const float*)d_in[0];   // (512, 8192)
    const float* h     = (const float*)d_in[1];   // (512, 8192, 4)
    const float* gt    = (const float*)d_in[2];   // (512, 4)
    const float* point = (const float*)d_in[3];   // (512, 4096, 3)

    float* Mw      = (float*)d_ws;                // 512*9 floats
    float* partial = Mw + BATCH * 9;              // 512 floats

    gram_eig_kernel<<<BATCH, 256, 0, stream>>>(s, h, gt, Mw);
    point_loss_kernel<<<BATCH, 256, 0, stream>>>(point, Mw, partial);
    finalize_kernel<<<1, 512, 0, stream>>>(partial, (float*)d_out);
}
